// Decoder_17952963298116
// MI455X (gfx1250) — matrix operand order repaired and checked by reading, compiled, not run on hardware
//
#include <hip/hip_runtime.h>

typedef __attribute__((ext_vector_type(16))) _Float16 v16h;
typedef __attribute__((ext_vector_type(8)))  _Float16 v8h;
typedef __attribute__((ext_vector_type(4)))  _Float16 v4h;
typedef __attribute__((ext_vector_type(8)))  float    v8f;
typedef __attribute__((ext_vector_type(4)))  float    v4f;

#define CCH   768            // channels (in == out)
#define MTOT  65536          // TB * N = 64 * 1024
#define NSEQ  1024           // N tokens
#define BDIM  16             // batch (TB = T*B, T=4, B=16)
#define MTILE 64             // block M tile (all 8 waves share it)

// ---------------------------------------------------------------------------
// Kernel 0: convert W (f32, row-major [o][c]) -> f16, and zero stats buffer.
// ---------------------------------------------------------------------------
__global__ __launch_bounds__(256) void prep_kernel(const float* __restrict__ W,
                                                   _Float16* __restrict__ Wh,
                                                   float* __restrict__ sums) {
    int i = blockIdx.x * 256 + threadIdx.x;
    if (i < CCH * CCH) Wh[i] = (_Float16)W[i];
    if (i < 2 * CCH)   sums[i] = 0.0f;
}

// ---------------------------------------------------------------------------
// Kernel 1: GEMM y[m][o] = sum_c x[m][c] * W[o][c], f16 WMMA, f32 accum,
// y stored f16.  Block = 8 waves; block tile = 64(M) x 768(N):
//   - x streamed from HBM exactly once (block covers all 768 output chans)
//   - all waves share the M range -> W L2 traffic = 1024 blocks * 1.18MB
//   - LDS A tile double-buffered: one barrier per K-step
//   - B fragments double-buffered in registers
// ---------------------------------------------------------------------------
__global__ __launch_bounds__(256) void gemm_kernel(const float* __restrict__ x,
                                                   const _Float16* __restrict__ Wh,
                                                   _Float16* __restrict__ yh) {
    // 64 rows x 32 halves, row stride 40 halves (80B) -> conflict-free b128 reads
    __shared__ __attribute__((aligned(16))) _Float16 Alds[2][MTILE * 40];

    const int tid   = threadIdx.x;
    const int wave  = tid >> 5;
    const int lane  = tid & 31;
    const int lh    = lane & 15;        // N (or M) index within half-wave
    const int hi    = lane >> 4;        // half-wave select
    const int mbase = blockIdx.x * MTILE;
    const int nbase = wave * 96;

    const int arow = tid >> 3;          // 0..31 (thread stages rows arow, arow+32)
    const int acol = (tid & 7) * 4;     // 0..28

    const v8f vzero = {};
    v8f acc[4][6];
#pragma unroll
    for (int mt = 0; mt < 4; ++mt)
#pragma unroll
        for (int nt = 0; nt < 6; ++nt) acc[mt][nt] = vzero;

    // ---- prologue: stage K-chunk 0 into LDS buffer 0
    {
        const float* px0 = x + (size_t)(mbase + arow) * CCH + acol;
        const float* px1 = px0 + (size_t)32 * CCH;
        v4f xa0 = *(const v4f*)px0;
        v4f xa1 = *(const v4f*)px1;
        v4h h0, h1;
#pragma unroll
        for (int i = 0; i < 4; ++i) { h0[i] = (_Float16)xa0[i]; h1[i] = (_Float16)xa1[i]; }
        *(v4h*)(&Alds[0][arow * 40 + acol])        = h0;
        *(v4h*)(&Alds[0][(arow + 32) * 40 + acol]) = h1;
    }

    int p = 0;
    for (int kb = 0; kb < CCH; kb += 32, p ^= 1) {
        __syncthreads();

        // ---- stage next K-chunk into the other LDS buffer (overlaps WMMAs)
        if (kb + 32 < CCH) {
            const float* px0 = x + (size_t)(mbase + arow) * CCH + (kb + 32) + acol;
            const float* px1 = px0 + (size_t)32 * CCH;
            v4f xa0 = *(const v4f*)px0;
            v4f xa1 = *(const v4f*)px1;
            v4h h0, h1;
#pragma unroll
            for (int i = 0; i < 4; ++i) { h0[i] = (_Float16)xa0[i]; h1[i] = (_Float16)xa1[i]; }
            *(v4h*)(&Alds[p ^ 1][arow * 40 + acol])        = h0;
            *(v4h*)(&Alds[p ^ 1][(arow + 32) * 40 + acol]) = h1;
        }

        // ---- A fragments for the 4 M-tiles (ISA 16-bit A 16x32 layout)
        v16h afrag[4];
#pragma unroll
        for (int mt = 0; mt < 4; ++mt) {
            const _Float16* ab = &Alds[p][(mt * 16 + lh) * 40 + hi * 8];
            v8h a0 = *(const v8h*)ab;          // K = koff .. koff+7
            v8h a1 = *(const v8h*)(ab + 16);   // K = 16+koff .. 16+koff+7
#pragma unroll
            for (int i = 0; i < 8; ++i) { afrag[mt][i] = a0[i]; afrag[mt][8 + i] = a1[i]; }
        }

        // ---- B fragments (double-buffered) + 24 WMMAs
        v16h bfrag[2];
        {
            const _Float16* wb = Wh + (size_t)(nbase + lh) * CCH + kb + hi * 8;
            v8h b0 = *(const v8h*)wb;
            v8h b1 = *(const v8h*)(wb + 16);
#pragma unroll
            for (int i = 0; i < 8; ++i) { bfrag[0][i] = b0[i]; bfrag[0][8 + i] = b1[i]; }
        }
#pragma unroll
        for (int nt = 0; nt < 6; ++nt) {
            if (nt < 5) {
                const _Float16* wb = Wh + (size_t)(nbase + (nt + 1) * 16 + lh) * CCH + kb + hi * 8;
                v8h b0 = *(const v8h*)wb;
                v8h b1 = *(const v8h*)(wb + 16);
#pragma unroll
                for (int i = 0; i < 8; ++i) { bfrag[(nt + 1) & 1][i] = b0[i]; bfrag[(nt + 1) & 1][8 + i] = b1[i]; }
            }
#pragma unroll
            for (int mt = 0; mt < 4; ++mt)
                acc[mt][nt] = __builtin_amdgcn_wmma_f32_16x16x32_f16(
                    false, afrag[mt], false, bfrag[nt & 1], (short)0, acc[mt][nt], false, false);
        }
    }

    // ---- write y as f16 (C/D layout: VGPR r, lanes<16 -> M=r, lanes>=16 -> M=8+r)
#pragma unroll
    for (int mt = 0; mt < 4; ++mt)
#pragma unroll
        for (int nt = 0; nt < 6; ++nt) {
            int ch = nbase + nt * 16 + lh;
#pragma unroll
            for (int r = 0; r < 8; ++r) {
                int m = mbase + mt * 16 + hi * 8 + r;
                yh[(size_t)m * CCH + ch] = (_Float16)acc[mt][nt][r];
            }
        }
}

// ---------------------------------------------------------------------------
// Kernel 2: per-channel sum and sum-of-squares over all M rows.
// 256 blocks x 192 threads; block handles 256 rows; thread handles 4 channels.
// ---------------------------------------------------------------------------
__global__ __launch_bounds__(192) void stats_kernel(const _Float16* __restrict__ yh,
                                                    float* __restrict__ sums) {
    int c4 = threadIdx.x * 4;
    int rowbase = blockIdx.x * 256;
    v4f s = {}, q = {};
    for (int r = 0; r < 256; ++r) {
        v4h y4 = *(const v4h*)(yh + (size_t)(rowbase + r) * CCH + c4);
#pragma unroll
        for (int i = 0; i < 4; ++i) {
            float v = (float)y4[i];
            s[i] += v;
            q[i] += v * v;
        }
    }
#pragma unroll
    for (int i = 0; i < 4; ++i) {
        atomicAdd(&sums[c4 + i], s[i]);
        atomicAdd(&sums[CCH + c4 + i], q[i]);
    }
}

// ---------------------------------------------------------------------------
// Kernel 3: finalize BN -> per-channel scale/shift.
// ---------------------------------------------------------------------------
__global__ __launch_bounds__(256) void finalize_kernel(const float* __restrict__ sums,
                                                       const float* __restrict__ gamma,
                                                       const float* __restrict__ beta,
                                                       float* __restrict__ ss) {
    int c = blockIdx.x * 256 + threadIdx.x;
    if (c >= CCH) return;
    const float inv_n = 1.0f / (float)MTOT;
    float mean = sums[c] * inv_n;
    float var  = sums[CCH + c] * inv_n - mean * mean;
    float sc   = gamma[c] * rsqrtf(var + 1e-5f);
    ss[c]       = sc;
    ss[CCH + c] = beta[c] - mean * sc;
}

// ---------------------------------------------------------------------------
// Kernel 4: fused BN + LIF (tau=2, v_th=1, hard reset) + spike writeout.
// One thread per (b, n, 4 channels); loops T=4 steps. out[t*16+b, n, c].
// ---------------------------------------------------------------------------
__global__ __launch_bounds__(256) void lif_kernel(const _Float16* __restrict__ yh,
                                                  const float* __restrict__ ss,
                                                  float* __restrict__ out) {
    int tid = blockIdx.x * 256 + threadIdx.x;       // B*N*C/4 = 3,145,728 threads
    int cg = tid % 192;                             // channel group (4 channels)
    int bn = tid / 192;
    int b  = bn >> 10;                              // bn / 1024
    int n  = bn & 1023;
    int c4 = cg * 4;
    v4f sc = *(const v4f*)(ss + c4);
    v4f sh = *(const v4f*)(ss + CCH + c4);
    v4f v = {};
#pragma unroll
    for (int t = 0; t < 4; ++t) {
        size_t idx = ((size_t)((t * BDIM + b) * NSEQ + n)) * CCH + c4;
        v4h y4 = *(const v4h*)(yh + idx);
        v4f s;
#pragma unroll
        for (int i = 0; i < 4; ++i) {
            float xv = (float)y4[i] * sc[i] + sh[i];
            v[i] += (xv - v[i]) * 0.5f;             // v += (x - v)/tau
            s[i] = (v[i] >= 1.0f) ? 1.0f : 0.0f;    // Heaviside(v - v_th)
            v[i] *= (1.0f - s[i]);                  // hard reset to 0
        }
        *(v4f*)(out + idx) = s;                     // b128 spike store
    }
}

// ---------------------------------------------------------------------------
// Workspace layout (bytes):
//   [0,        1179648)  Wh   : 768*768 f16
//   [1179648,  1185792)  sums : 2*768 f32 (sum, sumsq)
//   [1185792,  1191936)  ss   : 2*768 f32 (scale, shift)
//   [1191936,  +96MB  )  yh   : 65536*768 f16
// ---------------------------------------------------------------------------
extern "C" void kernel_launch(void* const* d_in, const int* in_sizes, int n_in,
                              void* d_out, int out_size, void* d_ws, size_t ws_size,
                              hipStream_t stream) {
    const float* x     = (const float*)d_in[0];
    const float* W     = (const float*)d_in[1];
    const float* gamma = (const float*)d_in[2];
    const float* beta  = (const float*)d_in[3];
    // d_in[4] = T (== 4, baked in)
    float* out = (float*)d_out;

    char* ws = (char*)d_ws;
    _Float16* Wh   = (_Float16*)(ws);
    float*    sums = (float*)(ws + 1179648);
    float*    ss   = (float*)(ws + 1185792);
    _Float16* yh   = (_Float16*)(ws + 1191936);

    prep_kernel    <<<(CCH * CCH + 255) / 256, 256, 0, stream>>>(W, Wh, sums);
    gemm_kernel    <<<MTOT / MTILE,            256, 0, stream>>>(x, Wh, yh);
    stats_kernel   <<<MTOT / 256,              192, 0, stream>>>(yh, sums);
    finalize_kernel<<<3,                       256, 0, stream>>>(sums, gamma, beta, ss);
    lif_kernel     <<<(BDIM * NSEQ * CCH / 4) / 256, 256, 0, stream>>>(yh, ss, out);
}
